// Block_57947698757665
// MI455X (gfx1250) — compile-verified, matrix operand order checked by reading, not run on hardware
//
#include <hip/hip_runtime.h>
#include <hip/hip_bf16.h>

typedef __attribute__((ext_vector_type(16))) _Float16 v16h;
typedef __attribute__((ext_vector_type(8)))  _Float16 v8h;
typedef __attribute__((ext_vector_type(8)))  float    v8f;

union F16x16 { v16h v; v8h h[2]; };

#define WMMA_F16(a, b, c) \
  __builtin_amdgcn_wmma_f32_16x16x32_f16(false, (a), false, (b), (short)0, (c), false, false)

// ---------------------------------------------------------------------------
// CDNA5 async global->LDS copy (16B per lane), tracked by ASYNCcnt.
// LDS destination address = low 32 bits of the generic pointer (LDS aperture
// maps addr[31:0] to the LDS offset). GV addressing: 64-bit VGPR address.
// ---------------------------------------------------------------------------
__device__ __forceinline__ void async_copy_b128(void* lds_dst, const void* gsrc) {
  unsigned lds_off = (unsigned)(size_t)lds_dst;
  unsigned long long ga = (unsigned long long)(size_t)gsrc;
  asm volatile("global_load_async_to_lds_b128 %0, %1, off"
               :: "v"(lds_off), "v"(ga) : "memory");
}
__device__ __forceinline__ void async_wait0() {
  asm volatile("s_wait_asynccnt 0x0" ::: "memory");
}

// ---------------------------------------------------------------------------
// f32 -> f16 conversion (grid-stride)
// ---------------------------------------------------------------------------
__global__ void cvt_f32_f16(const float* __restrict__ in, _Float16* __restrict__ out, int n) {
  int i = blockIdx.x * blockDim.x + threadIdx.x;
  int stride = gridDim.x * blockDim.x;
  for (; i < n; i += stride) out[i] = (_Float16)in[i];
}

// ---------------------------------------------------------------------------
// WMMA GEMM: C[M,N] = act(A[M,K] @ B[K,N] + bias[N])
// A,B f16 row-major. Block 256 threads (8 waves), tile 128x128, K-step 32.
// Wave grid 2x4; each wave computes 64x32 (4x2 WMMA tiles).
// A tile staged via GLOBAL_LOAD_ASYNC_TO_LDS_B128; B tile staged transposed
// through VGPRs (transpose can't be expressed as a straight async copy).
// ---------------------------------------------------------------------------
#define BM 128
#define BN 128
#define BKK 32
#define LDSA (BKK + 8) // 40 halfs = 80B rows (16B-aligned chunks)

__global__ __launch_bounds__(256) void gemm_f16_wmma(
    const _Float16* __restrict__ A, const _Float16* __restrict__ Bm,
    const float* __restrict__ bias,
    float* __restrict__ Cf, _Float16* __restrict__ Ch,
    int M, int N, int K, int act)
{
  __shared__ _Float16 As[BM][LDSA];
  __shared__ _Float16 Bs[BN][LDSA]; // transposed: Bs[n][k]

  const int tid  = threadIdx.x;
  const int lane = tid & 31;
  const int wave = tid >> 5;
  const int wr   = wave >> 2;   // 0..1
  const int wc   = wave & 3;    // 0..3
  const int lr   = lane & 15;
  const int hi   = lane >> 4;   // 0/1
  const int kb   = hi * 8;      // K-chunk base per 16-bit A/B lane layout

  const int m0 = blockIdx.y * BM;
  const int n0 = blockIdx.x * BN;

  v8f acc[4][2] = {};

  for (int k0 = 0; k0 < K; k0 += BKK) {
    // Stage A tile asynchronously: 128x32 halfs = 512 chunks of 16B
#pragma unroll
    for (int i = 0; i < 2; ++i) {
      int c = tid + i * 256;
      int r = c >> 2;
      int kc = (c & 3) * 8;
      async_copy_b128(&As[r][kc], &A[(size_t)(m0 + r) * K + k0 + kc]);
    }
    // Stage B tile transposed: 32x128 -> Bs[n][k]
#pragma unroll
    for (int i = 0; i < 2; ++i) {
      int c = tid + i * 256;
      int kk = c >> 4;
      int nn = (c & 15) * 8;
      v8h val = *reinterpret_cast<const v8h*>(&Bm[(size_t)(k0 + kk) * N + n0 + nn]);
#pragma unroll
      for (int j = 0; j < 8; ++j) Bs[nn + j][kk] = val[j];
    }
    async_wait0();
    __syncthreads();

    F16x16 af[4], bf[2];
#pragma unroll
    for (int i = 0; i < 4; ++i) {
      const _Float16* p = &As[wr * 64 + i * 16 + lr][0];
      af[i].h[0] = *reinterpret_cast<const v8h*>(p + kb);
      af[i].h[1] = *reinterpret_cast<const v8h*>(p + kb + 16);
    }
#pragma unroll
    for (int j = 0; j < 2; ++j) {
      const _Float16* p = &Bs[wc * 32 + j * 16 + lr][0];
      bf[j].h[0] = *reinterpret_cast<const v8h*>(p + kb);
      bf[j].h[1] = *reinterpret_cast<const v8h*>(p + kb + 16);
    }
#pragma unroll
    for (int i = 0; i < 4; ++i)
#pragma unroll
      for (int j = 0; j < 2; ++j)
        acc[i][j] = WMMA_F16(af[i].v, bf[j].v, acc[i][j]);
    __syncthreads();
  }

  // Epilogue: D layout = lane(n), vgpr v -> row v + 8*hi
#pragma unroll
  for (int i = 0; i < 4; ++i) {
#pragma unroll
    for (int j = 0; j < 2; ++j) {
      int gn = n0 + wc * 32 + j * 16 + lr;
      float bv = bias[gn];
#pragma unroll
      for (int v = 0; v < 8; ++v) {
        int gm = m0 + wr * 64 + i * 16 + v + hi * 8;
        float x = acc[i][j][v] + bv;
        if (act) x = 0.5f * x * (1.0f + erff(x * 0.70710678118654752f)); // exact GELU
        size_t idx = (size_t)gm * N + gn;
        if (Cf) Cf[idx] = x;
        if (Ch) Ch[idx] = (_Float16)x;
      }
    }
  }
}

// ---------------------------------------------------------------------------
// Flash attention over f16 qkv: qkv layout (B, T, 3C) with [K | Q | V] thirds.
// Block = (batch*head, q-tile of 64 rows). 128 threads = 4 waves, 16 rows each.
// HD = 64. Causal. Output y (f16) in (B, T, C) layout at head offset.
// K tile staged via async copy; V tile staged transposed through VGPRs.
// ---------------------------------------------------------------------------
__global__ __launch_bounds__(128) void attn_flash_wmma(
    const _Float16* __restrict__ qkvh, _Float16* __restrict__ Yh,
    int Bx, int Tx, int Cx, int NHx)
{
  __shared__ _Float16 Ks[64][72];      // key-major (rows = key, cols = d)
  __shared__ _Float16 Vt[64][72];      // d-major   (rows = d,   cols = key)
  __shared__ _Float16 Pl[4][16][72];   // wave-private P scratch

  const int tid  = threadIdx.x;
  const int lane = tid & 31;
  const int w    = tid >> 5;   // 0..3
  const int lr   = lane & 15;
  const int hi   = lane >> 4;
  const int kb   = hi * 8;

  const int bh = blockIdx.x;
  const int b  = bh / NHx;
  const int h  = bh % NHx;
  const int qb = blockIdx.y;
  const int q0 = qb * 64;

  const size_t rstride = (size_t)3 * Cx;
  const _Float16* kbase = qkvh + (size_t)b * Tx * rstride + 0 * Cx + h * 64;
  const _Float16* qbase = qkvh + (size_t)b * Tx * rstride + 1 * Cx + h * 64;
  const _Float16* vbase = qkvh + (size_t)b * Tx * rstride + 2 * Cx + h * 64;

  // Q fragments for this wave's 16 rows, pre-scaled by 1/sqrt(64) = 0.125
  F16x16 qf[2];
  {
    const int qrow = q0 + w * 16 + lr;
#pragma unroll
    for (int kk = 0; kk < 2; ++kk) {
      const _Float16* p = qbase + (size_t)qrow * rstride + kk * 32;
      v8h x0 = *reinterpret_cast<const v8h*>(p + kb);
      v8h x1 = *reinterpret_cast<const v8h*>(p + kb + 16);
#pragma unroll
      for (int j = 0; j < 8; ++j) {
        x0[j] = (_Float16)(x0[j] * (_Float16)0.125f);
        x1[j] = (_Float16)(x1[j] * (_Float16)0.125f);
      }
      qf[kk].h[0] = x0; qf[kk].h[1] = x1;
    }
  }

  float mrow[8], lsum[8];
  v8f O[4] = {};
#pragma unroll
  for (int v = 0; v < 8; ++v) { mrow[v] = -3.0e38f; lsum[v] = 0.0f; }

  for (int jb = 0; jb <= qb; ++jb) {
    const int key0 = jb * 64;
    // K tile: async straight copy (key-major). V tile: transpose via VGPRs.
#pragma unroll
    for (int i = 0; i < 4; ++i) {
      int c = tid + i * 128;
      int key = c >> 3;
      int dc  = (c & 7) * 8;
      async_copy_b128(&Ks[key][dc], kbase + (size_t)(key0 + key) * rstride + dc);
      v8h vv = *reinterpret_cast<const v8h*>(vbase + (size_t)(key0 + key) * rstride + dc);
#pragma unroll
      for (int j = 0; j < 8; ++j) Vt[dc + j][key] = vv[j];
    }
    async_wait0();
    __syncthreads();

    // S = (Q/8) @ K^T : 4 n-tiles of 16 keys, K-dim 64 = 2 WMMAs each
    v8f S[4];
#pragma unroll
    for (int nt = 0; nt < 4; ++nt) {
      S[nt] = (v8f){};
#pragma unroll
      for (int kk = 0; kk < 2; ++kk) {
        F16x16 bk;
        const _Float16* p = &Ks[nt * 16 + lr][kk * 32];
        bk.h[0] = *reinterpret_cast<const v8h*>(p + kb);
        bk.h[1] = *reinterpret_cast<const v8h*>(p + kb + 16);
        S[nt] = WMMA_F16(qf[kk].v, bk.v, S[nt]);
      }
    }
    // Causal mask (only the diagonal block needs it)
    if (jb == qb) {
#pragma unroll
      for (int nt = 0; nt < 4; ++nt) {
        int col = key0 + nt * 16 + lr;
#pragma unroll
        for (int v = 0; v < 8; ++v) {
          int row = q0 + w * 16 + v + hi * 8;
          if (col > row) S[nt][v] = -3.0e38f;
        }
      }
    }

    // Online softmax update per row (row m = v + 8*hi lives in one 16-lane half)
#pragma unroll
    for (int v = 0; v < 8; ++v) {
      float t = S[0][v];
#pragma unroll
      for (int nt = 1; nt < 4; ++nt) t = fmaxf(t, S[nt][v]);
#pragma unroll
      for (int off = 1; off < 16; off <<= 1) t = fmaxf(t, __shfl_xor(t, off, 32));
      float mn = fmaxf(mrow[v], t);
      float alpha = __expf(mrow[v] - mn);
      mrow[v] = mn;
      lsum[v] *= alpha;
#pragma unroll
      for (int ot = 0; ot < 4; ++ot) O[ot][v] *= alpha;
      float rs = 0.0f;
#pragma unroll
      for (int nt = 0; nt < 4; ++nt) {
        float p = __expf(S[nt][v] - mn);
        rs += p;
        Pl[w][v + hi * 8][nt * 16 + lr] = (_Float16)p;
      }
#pragma unroll
      for (int off = 1; off < 16; off <<= 1) rs += __shfl_xor(rs, off, 32);
      lsum[v] += rs;
    }

    // O += P @ V (P via wave-private LDS re-layout into A fragments)
#pragma unroll
    for (int ot = 0; ot < 4; ++ot) {
#pragma unroll
      for (int kk = 0; kk < 2; ++kk) {
        F16x16 ap, bv;
        const _Float16* pp = &Pl[w][lr][kk * 32];
        ap.h[0] = *reinterpret_cast<const v8h*>(pp + kb);
        ap.h[1] = *reinterpret_cast<const v8h*>(pp + kb + 16);
        const _Float16* vp = &Vt[ot * 16 + lr][kk * 32];
        bv.h[0] = *reinterpret_cast<const v8h*>(vp + kb);
        bv.h[1] = *reinterpret_cast<const v8h*>(vp + kb + 16);
        O[ot] = WMMA_F16(ap.v, bv.v, O[ot]);
      }
    }
    __syncthreads();
  }

  // Write y (f16), normalized
#pragma unroll
  for (int ot = 0; ot < 4; ++ot) {
#pragma unroll
    for (int v = 0; v < 8; ++v) {
      int r = q0 + w * 16 + v + hi * 8;
      int cidx = h * 64 + ot * 16 + lr;
      Yh[((size_t)b * Tx + r) * Cx + cidx] = (_Float16)(O[ot][v] / lsum[v]);
    }
  }
}

// ---------------------------------------------------------------------------
// out = xin + LayerNorm(t)*g + b  (row length C). One block per row.
// ---------------------------------------------------------------------------
__global__ __launch_bounds__(256) void ln_residual(
    const float* __restrict__ xin, const float* __restrict__ t,
    const float* __restrict__ g, const float* __restrict__ be,
    float* __restrict__ outf, _Float16* __restrict__ outh, int Cx)
{
  const int row = blockIdx.x;
  const int tid = threadIdx.x;
  const float* tr = t + (size_t)row * Cx;

  float s = 0.0f, s2 = 0.0f;
  for (int i = tid; i < Cx; i += 256) { float v = tr[i]; s += v; s2 += v * v; }
  __shared__ float shs[8], shs2[8];
#pragma unroll
  for (int off = 16; off; off >>= 1) { s += __shfl_xor(s, off, 32); s2 += __shfl_xor(s2, off, 32); }
  if ((tid & 31) == 0) { shs[tid >> 5] = s; shs2[tid >> 5] = s2; }
  __syncthreads();
  s = 0.0f; s2 = 0.0f;
#pragma unroll
  for (int i = 0; i < 8; ++i) { s += shs[i]; s2 += shs2[i]; }
  float mean = s / Cx;
  float var  = s2 / Cx - mean * mean;
  float rstd = rsqrtf(var + 1e-5f);

  for (int i = tid; i < Cx; i += 256) {
    float v = xin[(size_t)row * Cx + i] + (tr[i] - mean) * rstd * g[i] + be[i];
    if (outf) outf[(size_t)row * Cx + i] = v;
    if (outh) outh[(size_t)row * Cx + i] = (_Float16)v;
  }
}

// ---------------------------------------------------------------------------
extern "C" void kernel_launch(void* const* d_in, const int* in_sizes, int n_in,
                              void* d_out, int out_size, void* d_ws, size_t ws_size,
                              hipStream_t stream) {
  (void)in_sizes; (void)n_in; (void)out_size; (void)ws_size;
  const int Bb = 2, T = 2048, C = 1024, NH = 16;
  const int M = Bb * T; // 4096

  const float* x      = (const float*)d_in[0];
  const float* w_attn = (const float*)d_in[1];
  const float* b_attn = (const float*)d_in[2];
  const float* wa1    = (const float*)d_in[3];
  const float* ba1    = (const float*)d_in[4];
  const float* wa2    = (const float*)d_in[5];
  const float* ba2    = (const float*)d_in[6];
  const float* g1     = (const float*)d_in[7];
  const float* be1    = (const float*)d_in[8];
  const float* wf1    = (const float*)d_in[9];
  const float* bf1    = (const float*)d_in[10];
  const float* wf2    = (const float*)d_in[11];
  const float* bf2    = (const float*)d_in[12];
  const float* g2     = (const float*)d_in[13];
  const float* be2    = (const float*)d_in[14];
  float* out = (float*)d_out;

  char* ws = (char*)d_ws;
  size_t off = 0;
  auto take = [&](size_t bytes) -> char* {
    char* p = ws + off;
    off += (bytes + 255) & ~(size_t)255;
    return p;
  };

  _Float16* Xh    = (_Float16*)take((size_t)M * C * 2);        // x f16; reused as Yh
  _Float16* Wattn = (_Float16*)take((size_t)C * 3 * C * 2);
  _Float16* Wa1h  = (_Float16*)take((size_t)C * 4 * C * 2);
  _Float16* Wa2h  = (_Float16*)take((size_t)4 * C * C * 2);
  _Float16* Wf1h  = (_Float16*)take((size_t)C * 4 * C * 2);
  _Float16* Wf2h  = (_Float16*)take((size_t)4 * C * C * 2);
  char*     qkvB  = take((size_t)M * 3 * C * 2);               // qkv f16; reused as t2/t4 f32
  _Float16* H1h   = (_Float16*)take((size_t)M * 4 * C * 2);    // FF hidden; reused for both FFs
  float*    x1    = (float*)take((size_t)M * C * 4);
  _Float16* x1h   = (_Float16*)take((size_t)M * C * 2);

  _Float16* qkvh = (_Float16*)qkvB;
  float*    t2f  = (float*)qkvB;   // reuse after attention consumes qkv
  _Float16* Yh   = Xh;             // reuse after QKV GEMM consumes Xh

  // f32 -> f16 conversions
  cvt_f32_f16<<<4096, 256, 0, stream>>>(x,      Xh,    M * C);
  cvt_f32_f16<<<4096, 256, 0, stream>>>(w_attn, Wattn, C * 3 * C);
  cvt_f32_f16<<<4096, 256, 0, stream>>>(wa1,    Wa1h,  C * 4 * C);
  cvt_f32_f16<<<4096, 256, 0, stream>>>(wa2,    Wa2h,  4 * C * C);
  cvt_f32_f16<<<4096, 256, 0, stream>>>(wf1,    Wf1h,  C * 4 * C);
  cvt_f32_f16<<<4096, 256, 0, stream>>>(wf2,    Wf2h,  4 * C * C);

  // qkv = x @ w_attn + b_attn  -> f16
  gemm_f16_wmma<<<dim3(3 * C / BN, M / BM), 256, 0, stream>>>(
      Xh, Wattn, b_attn, nullptr, qkvh, M, 3 * C, C, 0);

  // flash attention -> Yh (f16)
  attn_flash_wmma<<<dim3(Bb * NH, T / 64), 128, 0, stream>>>(qkvh, Yh, Bb, T, C, NH);

  // SelfAttn FF: H1 = gelu(y @ wa1 + ba1); t2 = H1 @ wa2 + ba2
  gemm_f16_wmma<<<dim3(4 * C / BN, M / BM), 256, 0, stream>>>(
      Yh, Wa1h, ba1, nullptr, H1h, M, 4 * C, C, 1);
  gemm_f16_wmma<<<dim3(C / BN, M / BM), 256, 0, stream>>>(
      H1h, Wa2h, ba2, t2f, nullptr, M, C, 4 * C, 0);

  // x1 = x + LN(t2)
  ln_residual<<<M, 256, 0, stream>>>(x, t2f, g1, be1, x1, x1h, C);

  // Block FF: H2 = gelu(x1 @ wf1 + bf1); t4 = H2 @ wf2 + bf2
  gemm_f16_wmma<<<dim3(4 * C / BN, M / BM), 256, 0, stream>>>(
      x1h, Wf1h, bf1, nullptr, H1h, M, 4 * C, C, 1);
  gemm_f16_wmma<<<dim3(C / BN, M / BM), 256, 0, stream>>>(
      H1h, Wf2h, bf2, t2f, nullptr, M, C, 4 * C, 0);

  // out = x1 + LN(t4)
  ln_residual<<<M, 256, 0, stream>>>(x1, t2f, g2, be2, out, nullptr, C);
}
